// Attention_14044543058417
// MI455X (gfx1250) — compile-verified, matrix operand order checked by reading, not run on hardware
//
#include <hip/hip_runtime.h>
#include <hip/hip_bf16.h>

// ---------------- problem constants ----------------
#define BB     2
#define SS     2048
#define DIM    2048
#define NH     16
#define NKV    8
#define HD     128
#define HALF   64
#define NGRP   (NH / NKV)
#define SCALE  0.08838834764831845f   // 1/sqrt(128)

typedef __attribute__((ext_vector_type(16))) __bf16 v16bf;
typedef __attribute__((ext_vector_type(8)))  float  v8f;
typedef __attribute__((ext_vector_type(4)))  unsigned u32x4;

// ---------------- bf16 helpers (bit-exact, no libcalls) ----------------
__device__ __forceinline__ __bf16 f2bf(float f) {
    union { float f; unsigned u; } v; v.f = f;
    unsigned r = v.u + 0x7FFFu + ((v.u >> 16) & 1u);   // round-to-nearest-even
    unsigned short h = (unsigned short)(r >> 16);
    return __builtin_bit_cast(__bf16, h);
}
__device__ __forceinline__ float bf2f(__bf16 b) {
    unsigned short h = __builtin_bit_cast(unsigned short, b);
    unsigned u = ((unsigned)h) << 16;
    return __builtin_bit_cast(float, u);
}

// ---------------- WMMA fragment loads ----------------
// 16-bit A/B fragment (16x32 over K), ISA 7.12.2 layout:
//   lane L: row/col = L%16 ; half = L/16
//   VGPR 0..3 : K = {0..7}  + 8*half   (elements 0..7  from p)
//   VGPR 4..7 : K = {16..23}+ 8*half   (elements 16..23 from p)
// Both halves are 16 contiguous bytes -> two b128 loads (second at imm offset 32B).
__device__ __forceinline__ v16bf load_frag_p(const __bf16* p) {
    union { v16bf v; u32x4 q[2]; } f;
    f.q[0] = *(const u32x4*)(p);
    f.q[1] = *(const u32x4*)(p + 16);
    return f.v;
}
// convenience form: tile base + leading dim + k offset
__device__ __forceinline__ v16bf load_frag(const __bf16* tile, long ld, int lane, int koff) {
    int g = (lane >> 4) & 1;
    int r = lane & 15;
    return load_frag_p(tile + (long)r * ld + koff + 8 * g);
}

__device__ __forceinline__ v8f wmma_bf16(v16bf a, v16bf b, v8f c) {
    return __builtin_amdgcn_wmma_f32_16x16x32_bf16(false, a, false, b, (short)0, c, false, false);
}

// ---------------- elementwise / layout kernels ----------------
__global__ void conv_bf16_kernel(const float* __restrict__ in, __bf16* __restrict__ out, long n) {
    long i = (long)blockIdx.x * blockDim.x + threadIdx.x;
    if (i < n) out[i] = f2bf(in[i]);
}

// W [K][N] f32  ->  Wt [N][K] bf16
__global__ void transpose_conv_kernel(const float* __restrict__ W, __bf16* __restrict__ Wt,
                                      int K, int N) {
    long i = (long)blockIdx.x * blockDim.x + threadIdx.x;
    if (i >= (long)K * N) return;
    int n = (int)(i % N);
    int k = (int)(i / N);
    Wt[(long)n * K + k] = f2bf(W[i]);
}

// RoPE + rearrange: in [B*S][nh*128] -> out [B][nh][S][128]
__global__ void rope_kernel(const __bf16* __restrict__ in,
                            const float* __restrict__ cosT, const float* __restrict__ sinT,
                            __bf16* __restrict__ out, int nh) {
    long i = (long)blockIdx.x * blockDim.x + threadIdx.x;   // pair index
    long total = (long)BB * SS * nh * HALF;
    if (i >= total) return;
    int f = (int)(i % HALF);
    int h = (int)((i / HALF) % nh);
    int s = (int)((i / ((long)HALF * nh)) % SS);
    int b = (int)(i / ((long)HALF * nh * SS));
    const __bf16* row = in + ((long)(b * SS + s)) * (nh * HD) + h * HD + 2 * f;
    float a = bf2f(row[0]), bb = bf2f(row[1]);
    float c = cosT[(long)s * HALF + f], sn = sinT[(long)s * HALF + f];
    __bf16* orow = out + (((long)(b * nh + h) * SS + s)) * HD + 2 * f;
    orow[0] = f2bf(a * c - bb * sn);
    orow[1] = f2bf(a * sn + bb * c);
}

// v0 [B*S][NKV*128] -> Vt [B][NKV][128][S]
__global__ void transpose_v_kernel(const __bf16* __restrict__ v0, __bf16* __restrict__ Vt) {
    long i = (long)blockIdx.x * blockDim.x + threadIdx.x;
    long total = (long)BB * SS * NKV * HD;
    if (i >= total) return;
    int d   = (int)(i % HD);
    int kvh = (int)((i / HD) % NKV);
    int s   = (int)((i / ((long)HD * NKV)) % SS);
    int b   = (int)(i / ((long)HD * NKV * SS));
    Vt[(((long)(b * NKV + kvh)) * HD + d) * SS + s] = v0[i];
}

// ---------------- WMMA GEMM:  C[M,N] = A[M,K] * B[K,N],  Bt = B^T row-major ----------------
// block = 128 threads (4 waves); wave computes a 32x64 tile (2 A-frags x 4 B-frags,
// 8 WMMAs per 32-deep k-step). Full register double-buffering, branch-free rotated
// tail; launch_bounds(128,1) gives the allocator the whole register file.
__global__ void __launch_bounds__(128, 1)
gemm_wmma_kernel(const __bf16* __restrict__ A, const __bf16* __restrict__ Bt,
                 void* __restrict__ Cp, int M, int N, int K, int out_f32) {
    int lane = threadIdx.x & 31;
    int wave = threadIdx.x >> 5;
    int g = (lane >> 4) & 1;
    int r = lane & 15;
    int m0 = blockIdx.y * 32;
    int n0 = blockIdx.x * 256 + wave * 64;

    long lrow = (long)r * K + 8 * g;       // lane offset within a 16-row panel
    const __bf16* pA0 = A  + (long)m0 * K        + lrow;
    const __bf16* pA1 = A  + (long)(m0 + 16) * K + lrow;
    const __bf16* pB0 = Bt + (long)(n0 +  0) * K + lrow;
    const __bf16* pB1 = Bt + (long)(n0 + 16) * K + lrow;
    const __bf16* pB2 = Bt + (long)(n0 + 32) * K + lrow;
    const __bf16* pB3 = Bt + (long)(n0 + 48) * K + lrow;

    v8f acc0[4] = {};
    v8f acc1[4] = {};

    // prologue: fragments for k=0
    v16bf a0 = load_frag_p(pA0);
    v16bf a1 = load_frag_p(pA1);
    v16bf b0 = load_frag_p(pB0);
    v16bf b1 = load_frag_p(pB1);
    v16bf b2 = load_frag_p(pB2);
    v16bf b3 = load_frag_p(pB3);

    for (int k0 = 32; k0 < K; k0 += 32) {
        pA0 += 32; pA1 += 32; pB0 += 32; pB1 += 32; pB2 += 32; pB3 += 32;
        // next-step loads issued before this step's WMMAs -> overlap
        v16bf na0 = load_frag_p(pA0);
        v16bf na1 = load_frag_p(pA1);
        v16bf nb0 = load_frag_p(pB0);
        v16bf nb1 = load_frag_p(pB1);
        v16bf nb2 = load_frag_p(pB2);
        v16bf nb3 = load_frag_p(pB3);
        // L2 prefetch one further step ahead (global_prefetch_b8)
        __builtin_prefetch(pA0 + 32, 0, 3);
        __builtin_prefetch(pB0 + 32, 0, 3);

        acc0[0] = wmma_bf16(a0, b0, acc0[0]);
        acc1[0] = wmma_bf16(a1, b0, acc1[0]);
        acc0[1] = wmma_bf16(a0, b1, acc0[1]);
        acc1[1] = wmma_bf16(a1, b1, acc1[1]);
        acc0[2] = wmma_bf16(a0, b2, acc0[2]);
        acc1[2] = wmma_bf16(a1, b2, acc1[2]);
        acc0[3] = wmma_bf16(a0, b3, acc0[3]);
        acc1[3] = wmma_bf16(a1, b3, acc1[3]);

        a0 = na0; a1 = na1;
        b0 = nb0; b1 = nb1; b2 = nb2; b3 = nb3;
    }
    // tail k-step
    acc0[0] = wmma_bf16(a0, b0, acc0[0]);
    acc1[0] = wmma_bf16(a1, b0, acc1[0]);
    acc0[1] = wmma_bf16(a0, b1, acc0[1]);
    acc1[1] = wmma_bf16(a1, b1, acc1[1]);
    acc0[2] = wmma_bf16(a0, b2, acc0[2]);
    acc1[2] = wmma_bf16(a1, b2, acc1[2]);
    acc0[3] = wmma_bf16(a0, b3, acc0[3]);
    acc1[3] = wmma_bf16(a1, b3, acc1[3]);

    if (out_f32) {
        float* C = (float*)Cp;
#pragma unroll
        for (int t = 0; t < 4; ++t)
#pragma unroll
            for (int j = 0; j < 8; ++j) {
                C[(long)(m0 +      j + 8 * g) * N + n0 + t * 16 + r] = acc0[t][j];
                C[(long)(m0 + 16 + j + 8 * g) * N + n0 + t * 16 + r] = acc1[t][j];
            }
    } else {
        __bf16* C = (__bf16*)Cp;
#pragma unroll
        for (int t = 0; t < 4; ++t)
#pragma unroll
            for (int j = 0; j < 8; ++j) {
                C[(long)(m0 +      j + 8 * g) * N + n0 + t * 16 + r] = f2bf(acc0[t][j]);
                C[(long)(m0 + 16 + j + 8 * g) * N + n0 + t * 16 + r] = f2bf(acc1[t][j]);
            }
    }
}

// ---------------- Flash attention (causal, GQA) ----------------
// Qb [B][NH][S][128], Kb [B][NKV][S][128], Vt [B][NKV][128][S]
// attn out: [B*S][NH*128] bf16
// grid (S/64, NH, B), block 128 (4 waves, one 16-query tile each)
__global__ void __launch_bounds__(128, 1)
flash_kernel(const __bf16* __restrict__ Qb, const __bf16* __restrict__ Kb,
             const __bf16* __restrict__ Vt, __bf16* __restrict__ attn) {
    __shared__ __bf16 lds_p[4][16][32];   // per-wave 16x32 P tile
    int lane = threadIdx.x & 31;
    int wave = threadIdx.x >> 5;
    int g = (lane >> 4) & 1;
    int r = lane & 15;
    int h = blockIdx.y, b = blockIdx.z;
    int kvh = h / NGRP;
    int qbase = (blockIdx.x * 4 + wave) * 16;

    const __bf16* qptr = Qb + (((long)(b * NH + h)) * SS + qbase) * HD;
    const __bf16* kptr = Kb + ((long)(b * NKV + kvh)) * SS * HD;
    const __bf16* vptr = Vt + ((long)(b * NKV + kvh)) * HD * SS;

    v16bf aq[4];
#pragma unroll
    for (int d = 0; d < 4; ++d) aq[d] = load_frag(qptr, HD, lane, d * 32);

    v8f o[8] = {};
    float mrun[8], lrun[8];
#pragma unroll
    for (int j = 0; j < 8; ++j) { mrun[j] = -3.0e38f; lrun[j] = 0.0f; }

    int kend = qbase + 16;
    for (int kb = 0; kb < kend; kb += 32) {
        // prefetch next key/value tiles (global_prefetch_b8)
        if (kb + 32 < kend) {
            __builtin_prefetch(kptr + (long)(kb + 32 + r) * HD, 0, 3);
            __builtin_prefetch(vptr + (long)r * SS + kb + 32, 0, 3);
        }
        // ---- S = Q*K^T, K fragments rotated through a 2-deep buffer ----
        v8f s0 = {}, s1 = {};
        v16bf bk0 = load_frag(kptr + (long)kb * HD,        HD, lane, 0);
        v16bf bk1 = load_frag(kptr + (long)(kb + 16) * HD, HD, lane, 0);
#pragma unroll
        for (int d = 0; d < 4; ++d) {
            v16bf nk0 = bk0, nk1 = bk1;
            if (d < 3) {
                nk0 = load_frag(kptr + (long)kb * HD,        HD, lane, (d + 1) * 32);
                nk1 = load_frag(kptr + (long)(kb + 16) * HD, HD, lane, (d + 1) * 32);
            }
            s0 = wmma_bf16(aq[d], bk0, s0);
            s1 = wmma_bf16(aq[d], bk1, s1);
            bk0 = nk0; bk1 = nk1;
        }
        // ---- online softmax over the 32 new columns ----
        int k0c = kb + r, k1c = kb + 16 + r;
#pragma unroll
        for (int j = 0; j < 8; ++j) {
            int qrow = qbase + j + 8 * g;
            float v0 = (k0c <= qrow) ? s0[j] * SCALE : -3.0e38f;
            float v1 = (k1c <= qrow) ? s1[j] * SCALE : -3.0e38f;
            float rm = fmaxf(v0, v1);
#pragma unroll
            for (int off = 1; off < 16; off <<= 1)
                rm = fmaxf(rm, __shfl_xor(rm, off, 32));
            float mnew = fmaxf(mrun[j], rm);
            float corr = __expf(mrun[j] - mnew);
            float e0 = __expf(v0 - mnew);
            float e1 = __expf(v1 - mnew);
            float rs = e0 + e1;
#pragma unroll
            for (int off = 1; off < 16; off <<= 1)
                rs += __shfl_xor(rs, off, 32);
            lrun[j] = lrun[j] * corr + rs;
            mrun[j] = mnew;
#pragma unroll
            for (int dt = 0; dt < 8; ++dt) o[dt][j] *= corr;
            lds_p[wave][j + 8 * g][r]      = f2bf(e0);
            lds_p[wave][j + 8 * g][16 + r] = f2bf(e1);
        }
        // ---- O += P * V, V fragments rotated through a 2-deep buffer ----
        v16bf ap = load_frag(&lds_p[wave][0][0], 32, lane, 0);
        v16bf bv = load_frag(vptr + 0, SS, lane, kb);
#pragma unroll
        for (int dt = 0; dt < 8; ++dt) {
            v16bf nv = bv;
            if (dt < 7)
                nv = load_frag(vptr + (long)((dt + 1) * 16) * SS, SS, lane, kb);
            o[dt] = wmma_bf16(ap, bv, o[dt]);
            bv = nv;
        }
    }

    // ---- normalize and write [b*s][h*128+d] ----
#pragma unroll
    for (int j = 0; j < 8; ++j) {
        float inv = 1.0f / lrun[j];
        long row = (long)b * SS + qbase + j + 8 * g;
        __bf16* orow = attn + row * (NH * HD) + h * HD;
#pragma unroll
        for (int dt = 0; dt < 8; ++dt)
            orow[dt * 16 + r] = f2bf(o[dt][j] * inv);
    }
}

// ---------------- host-side orchestration ----------------
extern "C" void kernel_launch(void* const* d_in, const int* in_sizes, int n_in,
                              void* d_out, int out_size, void* d_ws, size_t ws_size,
                              hipStream_t stream) {
    (void)in_sizes; (void)n_in; (void)out_size; (void)ws_size;
    const float* x   = (const float*)d_in[0];
    const float* pco = (const float*)d_in[1];
    const float* psi = (const float*)d_in[2];
    const float* wq  = (const float*)d_in[3];
    const float* wk  = (const float*)d_in[4];
    const float* wv  = (const float*)d_in[5];
    const float* wo  = (const float*)d_in[6];
    float* out = (float*)d_out;

    const long M = (long)BB * SS;   // 4096
    char* ws = (char*)d_ws;
    auto take = [&](size_t bytes) { char* p = ws; ws += (bytes + 255) & ~(size_t)255; return p; };

    __bf16* xb  = (__bf16*)take(M * DIM * 2);
    __bf16* wqT = (__bf16*)take((size_t)DIM * DIM * 2);
    __bf16* wkT = (__bf16*)take((size_t)(NKV * HD) * DIM * 2);
    __bf16* wvT = (__bf16*)take((size_t)(NKV * HD) * DIM * 2);
    __bf16* woT = (__bf16*)take((size_t)DIM * DIM * 2);
    __bf16* q0  = (__bf16*)take(M * (NH  * HD) * 2);
    __bf16* k0  = (__bf16*)take(M * (NKV * HD) * 2);
    __bf16* v0  = (__bf16*)take(M * (NKV * HD) * 2);
    __bf16* Qb  = (__bf16*)take((size_t)BB * NH  * SS * HD * 2);
    __bf16* Kb  = (__bf16*)take((size_t)BB * NKV * SS * HD * 2);
    __bf16* Vt  = (__bf16*)take((size_t)BB * NKV * HD * SS * 2);
    __bf16* at  = (__bf16*)take(M * DIM * 2);

    // 1) x -> bf16
    {
        long n = M * DIM;
        conv_bf16_kernel<<<dim3((n + 255) / 256), dim3(256), 0, stream>>>(x, xb, n);
    }
    // 2) weight transposes -> bf16
    {
        long n = (long)DIM * DIM;
        transpose_conv_kernel<<<dim3((n + 255) / 256), dim3(256), 0, stream>>>(wq, wqT, DIM, DIM);
        transpose_conv_kernel<<<dim3((n + 255) / 256), dim3(256), 0, stream>>>(wo, woT, DIM, DIM);
        long nk = (long)DIM * (NKV * HD);
        transpose_conv_kernel<<<dim3((nk + 255) / 256), dim3(256), 0, stream>>>(wk, wkT, DIM, NKV * HD);
        transpose_conv_kernel<<<dim3((nk + 255) / 256), dim3(256), 0, stream>>>(wv, wvT, DIM, NKV * HD);
    }
    // 3) Q/K/V projections (WMMA)
    gemm_wmma_kernel<<<dim3(DIM / 256, M / 32), dim3(128), 0, stream>>>(xb, wqT, q0, (int)M, DIM, DIM, 0);
    gemm_wmma_kernel<<<dim3((NKV * HD) / 256, M / 32), dim3(128), 0, stream>>>(xb, wkT, k0, (int)M, NKV * HD, DIM, 0);
    gemm_wmma_kernel<<<dim3((NKV * HD) / 256, M / 32), dim3(128), 0, stream>>>(xb, wvT, v0, (int)M, NKV * HD, DIM, 0);
    // 4) RoPE + rearrange; V transpose
    {
        long nq = (long)BB * SS * NH * HALF;
        rope_kernel<<<dim3((nq + 255) / 256), dim3(256), 0, stream>>>(q0, pco, psi, Qb, NH);
        long nk = (long)BB * SS * NKV * HALF;
        rope_kernel<<<dim3((nk + 255) / 256), dim3(256), 0, stream>>>(k0, pco, psi, Kb, NKV);
        long nv = (long)BB * SS * NKV * HD;
        transpose_v_kernel<<<dim3((nv + 255) / 256), dim3(256), 0, stream>>>(v0, Vt);
    }
    // 5) flash attention (WMMA + online softmax)
    flash_kernel<<<dim3(SS / 64, NH, BB), dim3(128), 0, stream>>>(Qb, Kb, Vt, at);
    // 6) output projection -> f32
    gemm_wmma_kernel<<<dim3(DIM / 256, M / 32), dim3(128), 0, stream>>>(at, woT, out, (int)M, DIM, DIM, 1);
}
